// DensePlatonicAttention_86277303042489
// MI455X (gfx1250) — matrix operand order repaired and checked by reading, not compiled, not run on hardware
//
#include <hip/hip_runtime.h>
#include <hip/hip_bf16.h>

// Problem constants (match reference)
#define G_  12
#define H_  2
#define D_  64
#define C_  128   // CIN_G
#define E_  128   // EMB_G = H*D
#define O_  128   // OUT_G
#define F_  32    // D/2
#define SD_ 3
#define B_  4
#define N_  8192

typedef __attribute__((ext_vector_type(16))) __bf16 v16bf;
typedef __attribute__((ext_vector_type(8)))  float  v8f;

__device__ __forceinline__ v8f wmma_bf16(v16bf a, v16bf b, v8f c) {
  // D = A(16x32) * B(32x16) + C, f32 accumulate
  return __builtin_amdgcn_wmma_f32_16x16x32_bf16(false, a, false, b, (short)0, c,
                                                 false, false);
}

union BfPack8 { __bf16 h[8]; uint4 v; };
union BfPack4 { __bf16 h[4]; uint2 v; };

// ---- Fragment loaders (ISA 7.12.2 / 7.12.4 layouts, wave32) ------------------
// A fragment: 16(M) x 32(K) from a row-major [M][K] array.
// lane<16: row M=lane, halves K{0..7,16..23}; lane>=16: row M=lane-16, K{8..15,24..31}
__device__ __forceinline__ v16bf load_a_bf16(const __bf16* base, int stride, int lane) {
  int m   = lane & 15;
  int klo = (lane & 16) ? 8 : 0;
  const __bf16* p = base + m * stride + klo;
  v16bf a;
#pragma unroll
  for (int i = 0; i < 8; ++i) { a[i] = p[i]; a[i + 8] = p[i + 16]; }
  return a;
}

__device__ __forceinline__ v16bf load_a_f32(const float* base, int stride, int lane) {
  int m   = lane & 15;
  int klo = (lane & 16) ? 8 : 0;
  const float* p = base + m * stride + klo;
  v16bf a;
#pragma unroll
  for (int i = 0; i < 8; ++i) { a[i] = (__bf16)p[i]; a[i + 8] = (__bf16)p[i + 16]; }
  return a;
}

// B fragment: 32(K) x 16(N), sourced from B^T stored row-major as [N][K].
// lane<16: col N=lane holds K{0..7,16..23}; lane>=16: col N=lane-16, K{8..15,24..31}
__device__ __forceinline__ v16bf load_b_bf16(const __bf16* baseT, int stride, int lane) {
  int n   = lane & 15;
  int klo = (lane & 16) ? 8 : 0;
  const __bf16* p = baseT + n * stride + klo;
  v16bf b;
#pragma unroll
  for (int i = 0; i < 8; ++i) { b[i] = p[i]; b[i + 8] = p[i + 16]; }
  return b;
}

// Vectorized f32->bf16 x-tile stage: 32 tokens x 128 c into sX[t][c].
// 256 threads: lane-constant column group (tid&31)*4, rows tid>>5 + {0,8,16,24}
__device__ __forceinline__ void stage_x_tile(const float* __restrict__ xbase,
                                             __bf16* __restrict__ sX, int tid) {
  const int trow = tid >> 5;
  const int c4   = (tid & 31) * 4;
  const float* gp = xbase + (size_t)trow * (G_ * C_) + c4;
  __bf16* lp = sX + trow * C_ + c4;
#pragma unroll
  for (int j = 0; j < 4; ++j) {
    float4 v = *(const float4*)(gp + (size_t)j * 8 * (G_ * C_));
    BfPack4 p;
    p.h[0] = (__bf16)v.x; p.h[1] = (__bf16)v.y;
    p.h[2] = (__bf16)v.z; p.h[3] = (__bf16)v.w;
    *(uint2*)(lp + j * 8 * C_) = p.v;
  }
}

// ---- Kernel 0: weight conversion + kv accumulator zero ----------------------
__global__ __launch_bounds__(256) void prep_kernel(const float* __restrict__ Wq,
                                                   const float* __restrict__ Wv,
                                                   const float* __restrict__ Wo,
                                                   float* __restrict__ kv,
                                                   __bf16* __restrict__ wqb,
                                                   __bf16* __restrict__ wvb,
                                                   __bf16* __restrict__ wob) {
  int i = blockIdx.x * blockDim.x + threadIdx.x;
  int nt = gridDim.x * blockDim.x;
  if (i < E_ * C_) {
    wqb[i] = (__bf16)Wq[i];
    wvb[i] = (__bf16)Wv[i];
    wob[i] = (__bf16)Wo[i];
  }
  for (int j = i; j < B_ * G_ * H_ * D_ * D_; j += nt) kv[j] = 0.0f;
}

// ---- Kernel 1: kv[b,g,h] = (1/N) * K^T V  via WMMA + f32 atomics ------------
#define TS 256   // tokens per block (slice)
#define CK 32    // tokens per inner chunk (= WMMA K)

__global__ __launch_bounds__(256) void kv_kernel(const float* __restrict__ x,
                                                 const float* __restrict__ pos,
                                                 const __bf16* __restrict__ wvb,
                                                 const float* __restrict__ bv,
                                                 const float* __restrict__ freqs,
                                                 float* __restrict__ kv) {
  __shared__ __align__(16) __bf16 sWv[E_ * C_];    // [e][c]  32 KB
  __shared__ __align__(16) __bf16 sX[CK * C_];     // [t][c]   8 KB
  __shared__ __align__(16) __bf16 sVT[E_ * CK];    // [e][t]   8 KB (B^T for kv)
  __shared__ __align__(16) __bf16 sKT[E_ * CK];    // [d][t]   8 KB (A for kv)
  __shared__ float sC[CK * H_ * F_];               // cos      8 KB
  __shared__ float sS[CK * H_ * F_];               // sin      8 KB
  __shared__ float sPos[CK * SD_];
  __shared__ float sBv[E_];
  __shared__ float sFr[H_ * F_ * SD_];

  const int tid  = threadIdx.x;
  const int lane = tid & 31;
  const int wave = tid >> 5;
  const int slice = blockIdx.x, g = blockIdx.y, b = blockIdx.z;
  const int tok0 = slice * TS;

  // Stage Wv (bf16) via 16B vector copies
  {
    const uint4* src = (const uint4*)wvb;
    uint4* dst = (uint4*)sWv;
    for (int i = tid; i < (E_ * C_) / 8; i += 256) dst[i] = src[i];
  }
  if (tid < E_) sBv[tid] = bv[tid];
  if (tid < H_ * F_ * SD_) sFr[tid] = freqs[g * H_ * F_ * SD_ + tid];

  const int h_kv = wave >> 2;   // this wave's head for kv accumulation
  const int mt_kv = wave & 3;   // d1 tile
  v8f acc[4] = {};              // 4 N-tiles of kv, f32 accumulators

  for (int ch = 0; ch < TS / CK; ++ch) {
    const int t0 = tok0 + ch * CK;
    __syncthreads();  // previous iteration fully consumed
    if (ch + 1 < TS / CK && tid < 128) {
      // prefetch next chunk: one line per thread (32 rows x 4x128B lines)
      const float* nx = &x[((size_t)b * N_ + (t0 + CK) + (tid >> 2)) * (G_ * C_) +
                           g * C_ + (tid & 3) * 32];
      __builtin_prefetch(nx, 0, 3);
    }
    // Stage x chunk (f32 -> bf16), vectorized b128 loads
    stage_x_tile(&x[((size_t)b * N_ + t0) * (G_ * C_) + g * C_], sX, tid);
    // Stage pos tile
    if (tid < CK * SD_) sPos[tid] = pos[((size_t)b * N_ + t0) * SD_ + tid];
    __syncthreads();
    // Angles + cos/sin: per-thread fixed hf, rows t = (tid>>6) + 4j
    {
      const int hf = tid & 63;
      const float f0 = sFr[hf * SD_ + 0], f1 = sFr[hf * SD_ + 1], f2 = sFr[hf * SD_ + 2];
#pragma unroll
      for (int j = 0; j < 8; ++j) {
        int t = (tid >> 6) + 4 * j;
        float ang = sPos[t * SD_ + 0] * f0 + sPos[t * SD_ + 1] * f1 +
                    sPos[t * SD_ + 2] * f2;
        sC[t * (H_ * F_) + hf] = __cosf(ang);
        sS[t * (H_ * F_) + hf] = __sinf(ang);
      }
    }
    __syncthreads();
    // Roped K (k = rope(ones) = [cos - sin, cos + sin]) transposed: sKT[d][t]
    for (int i = tid; i < E_ * CK; i += 256) {
      int d = i >> 5, t = i & (CK - 1);
      int h = d >> 6, dd = d & 63, f = dd & 31;
      float c = sC[t * (H_ * F_) + h * F_ + f];
      float s = sS[t * (H_ * F_) + h * F_ + f];
      sKT[d * CK + t] = (__bf16)((dd < F_) ? (c - s) : (c + s));
    }
    // V projection: V(32xE) = X(32xC) @ Wv^T; 16 tiles over 8 waves, write V^T + bias
    for (int tt = wave; tt < 16; tt += 8) {
      int mt = tt & 1, nt = tt >> 1;
      v8f d = {};
#pragma unroll
      for (int ks = 0; ks < 4; ++ks) {
        v16bf a  = load_a_bf16(&sX[(mt * 16) * C_ + ks * 32], C_, lane);
        v16bf bb = load_b_bf16(&sWv[(nt * 16) * C_ + ks * 32], C_, lane);
        d = wmma_bf16(a, bb, d);
      }
      int e = nt * 16 + (lane & 15);
      float bvv = sBv[e];
      int mbase = mt * 16 + ((lane & 16) ? 8 : 0);
      BfPack8 p;
#pragma unroll
      for (int r = 0; r < 8; ++r) p.h[r] = (__bf16)(d[r] + bvv);
      *(uint4*)&sVT[e * CK + mbase] = p.v;   // 8 consecutive tokens, 16B store
    }
    __syncthreads();
    // kv accumulation: kv_h(64x64) += K_h^T(64x32) @ V_h(32x64)
    {
      v16bf a = load_a_bf16(&sKT[(h_kv * 64 + mt_kv * 16) * CK], CK, lane);
#pragma unroll
      for (int nt = 0; nt < 4; ++nt) {
        v16bf bb = load_b_bf16(&sVT[(h_kv * 64 + nt * 16) * CK], CK, lane);
        acc[nt] = wmma_bf16(a, bb, acc[nt]);
      }
    }
  }
  // Commit slice-partial kv with f32 global atomics (x 1/N)
  {
    const float scale = 1.0f / (float)N_;
    int n = lane & 15;
    int mbase = mt_kv * 16 + ((lane & 16) ? 8 : 0);
    float* kvh = &kv[(((size_t)(b * G_ + g) * H_ + h_kv) * D_) * D_];
#pragma unroll
    for (int nt = 0; nt < 4; ++nt)
#pragma unroll
      for (int r = 0; r < 8; ++r)
        atomicAdd(&kvh[(mbase + r) * D_ + nt * 16 + n], acc[nt][r] * scale);
  }
}

// ---- Kernel 2: W2^T[b,g][o][h*64+d1] = sum_d2 kv[b,g,h][d1,d2] * Wo[o,h*64+d2]
__global__ __launch_bounds__(128) void fold_kernel(const float* __restrict__ kv,
                                                   const __bf16* __restrict__ wob,
                                                   __bf16* __restrict__ w2t) {
  const int lane = threadIdx.x & 31;
  const int mt = threadIdx.x >> 5;     // d1 tile, 4 waves
  const int g = blockIdx.x, b = blockIdx.y;
  for (int h = 0; h < H_; ++h) {
    const float* kvh = &kv[(((size_t)(b * G_ + g) * H_ + h) * D_) * D_];
    for (int nt = 0; nt < 8; ++nt) {   // o tiles
      v8f d = {};
#pragma unroll
      for (int ks = 0; ks < 2; ++ks) { // d2 in 2 K-steps of 32
        v16bf a  = load_a_f32(kvh + (mt * 16) * D_ + ks * 32, D_, lane);
        v16bf bb = load_b_bf16(wob + (nt * 16) * E_ + h * 64 + ks * 32, E_, lane);
        d = wmma_bf16(a, bb, d);
      }
      int o = nt * 16 + (lane & 15);
      int mbase = mt * 16 + ((lane & 16) ? 8 : 0);
      BfPack8 p;
#pragma unroll
      for (int r = 0; r < 8; ++r) p.h[r] = (__bf16)d[r];
      *(uint4*)&w2t[((size_t)(b * G_ + g) * O_ + o) * E_ + h * 64 + mbase] = p.v;
    }
  }
}

// ---- Kernel 3: y = rope(x@Wq^T + bq) @ W2 + bo ------------------------------
__global__ __launch_bounds__(256) void out_kernel(const float* __restrict__ x,
                                                  const float* __restrict__ pos,
                                                  const __bf16* __restrict__ wqb,
                                                  const float* __restrict__ bq,
                                                  const float* __restrict__ freqs,
                                                  const __bf16* __restrict__ w2t,
                                                  const float* __restrict__ bo,
                                                  float* __restrict__ y) {
  __shared__ __align__(16) __bf16 sWq[E_ * C_];    // [e][c]  32 KB
  __shared__ __align__(16) __bf16 sW2[O_ * E_];    // [o][e]  32 KB (B^T)
  __shared__ __align__(16) __bf16 sX[32 * C_];     // [t][c]   8 KB
  __shared__ __align__(16) float  sQT[E_ * 32];    // [e][t]  16 KB (pre-rope, transposed)
  __shared__ __align__(16) __bf16 sQb[32 * E_];    // [t][e]   8 KB (post-rope A)
  __shared__ float sC[32 * H_ * F_];               //          8 KB
  __shared__ float sS[32 * H_ * F_];               //          8 KB
  __shared__ float sPos[32 * SD_];
  __shared__ float sBq[E_], sBo[O_], sFr[H_ * F_ * SD_];

  const int tid  = threadIdx.x;
  const int lane = tid & 31;
  const int wave = tid >> 5;
  const int tb = blockIdx.x, g = blockIdx.y, b = blockIdx.z;
  const int t0 = tb * 32;

  {
    const uint4* s0 = (const uint4*)wqb;
    const uint4* s1 = (const uint4*)(w2t + (size_t)(b * G_ + g) * O_ * E_);
    uint4* d0 = (uint4*)sWq;
    uint4* d1 = (uint4*)sW2;
    for (int i = tid; i < (E_ * C_) / 8; i += 256) { d0[i] = s0[i]; d1[i] = s1[i]; }
  }
  if (tid < E_) { sBq[tid] = bq[tid]; sBo[tid] = bo[tid]; }
  if (tid < H_ * F_ * SD_) sFr[tid] = freqs[g * H_ * F_ * SD_ + tid];
  if (tid < 32 * SD_) sPos[tid] = pos[((size_t)b * N_ + t0) * SD_ + tid];
  // Stage x tile (vectorized b128 loads)
  stage_x_tile(&x[((size_t)b * N_ + t0) * (G_ * C_) + g * C_], sX, tid);
  __syncthreads();
  // cos/sin
  {
    const int hf = tid & 63;
    const float f0 = sFr[hf * SD_ + 0], f1 = sFr[hf * SD_ + 1], f2 = sFr[hf * SD_ + 2];
#pragma unroll
    for (int j = 0; j < 8; ++j) {
      int t = (tid >> 6) + 4 * j;
      float ang = sPos[t * SD_ + 0] * f0 + sPos[t * SD_ + 1] * f1 +
                  sPos[t * SD_ + 2] * f2;
      sC[t * (H_ * F_) + hf] = __cosf(ang);
      sS[t * (H_ * F_) + hf] = __sinf(ang);
    }
  }
  __syncthreads();
  // Q projection: Q(32xE) = X @ Wq^T + bq, stored transposed sQT[e][t]
  for (int tt = wave; tt < 16; tt += 8) {
    int mt = tt & 1, nt = tt >> 1;
    v8f d = {};
#pragma unroll
    for (int ks = 0; ks < 4; ++ks) {
      v16bf a  = load_a_bf16(&sX[(mt * 16) * C_ + ks * 32], C_, lane);
      v16bf bb = load_b_bf16(&sWq[(nt * 16) * C_ + ks * 32], C_, lane);
      d = wmma_bf16(a, bb, d);
    }
    int e = nt * 16 + (lane & 15);
    float bqq = sBq[e];
    int mbase = mt * 16 + ((lane & 16) ? 8 : 0);
    float4 lo = make_float4(d[0] + bqq, d[1] + bqq, d[2] + bqq, d[3] + bqq);
    float4 hi = make_float4(d[4] + bqq, d[5] + bqq, d[6] + bqq, d[7] + bqq);
    *(float4*)&sQT[e * 32 + mbase]     = lo;   // 8 consecutive tokens: 2x b128
    *(float4*)&sQT[e * 32 + mbase + 4] = hi;
  }
  __syncthreads();
  // RoPE (f32 -> bf16 A-operand), pairs (e, e+32) within each head
  {
    const int hf = tid & 63, h = hf >> 5, f = hf & 31;
#pragma unroll
    for (int j = 0; j < 8; ++j) {
      int t = (tid >> 6) + 4 * j;
      float c = sC[t * (H_ * F_) + h * F_ + f];
      float s = sS[t * (H_ * F_) + h * F_ + f];
      float q1 = sQT[(h * 64 + f) * 32 + t];
      float q2 = sQT[(h * 64 + F_ + f) * 32 + t];
      sQb[t * E_ + h * 64 + f]      = (__bf16)(q1 * c - q2 * s);
      sQb[t * E_ + h * 64 + F_ + f] = (__bf16)(q1 * s + q2 * c);
    }
  }
  __syncthreads();
  // Output: Y(32xO) = Qr(32xE) @ W2 + bo
  for (int tt = wave; tt < 16; tt += 8) {
    int mt = tt & 1, nt = tt >> 1;
    v8f d = {};
#pragma unroll
    for (int ks = 0; ks < 4; ++ks) {
      v16bf a  = load_a_bf16(&sQb[(mt * 16) * E_ + ks * 32], E_, lane);
      v16bf bb = load_b_bf16(&sW2[(nt * 16) * E_ + ks * 32], E_, lane);
      d = wmma_bf16(a, bb, d);
    }
    int o = nt * 16 + (lane & 15);
    float boo = sBo[o];
    int mbase = mt * 16 + ((lane & 16) ? 8 : 0);
    float* yb = &y[((size_t)b * N_ + (t0 + mbase)) * (G_ * O_) + g * O_ + o];
#pragma unroll
    for (int r = 0; r < 8; ++r)
      yb[(size_t)r * (G_ * O_)] = d[r] + boo;   // coalesced b32 across lanes
  }
}

// ---- Host-side launcher -----------------------------------------------------
extern "C" void kernel_launch(void* const* d_in, const int* in_sizes, int n_in,
                              void* d_out, int out_size, void* d_ws, size_t ws_size,
                              hipStream_t stream) {
  const float* x     = (const float*)d_in[0];
  const float* pos   = (const float*)d_in[1];
  const float* Wq    = (const float*)d_in[2];
  const float* bq    = (const float*)d_in[3];
  const float* Wv    = (const float*)d_in[4];
  const float* bv    = (const float*)d_in[5];
  const float* Wo    = (const float*)d_in[6];
  const float* bo    = (const float*)d_in[7];
  const float* freqs = (const float*)d_in[8];
  float* y = (float*)d_out;

  // Workspace layout (16B-aligned sections)
  char* ws = (char*)d_ws;
  float*  kv  = (float*)(ws);                                  // 393216 f32
  __bf16* wqb = (__bf16*)(ws + 1572864);                       // 16384 bf16
  __bf16* wvb = (__bf16*)(ws + 1572864 + 32768);
  __bf16* wob = (__bf16*)(ws + 1572864 + 65536);
  __bf16* w2t = (__bf16*)(ws + 1572864 + 98304);               // 786432 bf16

  prep_kernel<<<192, 256, 0, stream>>>(Wq, Wv, Wo, kv, wqb, wvb, wob);
  kv_kernel<<<dim3(N_ / TS, G_, B_), 256, 0, stream>>>(x, pos, wvb, bv, freqs, kv);
  fold_kernel<<<dim3(G_, B_), 128, 0, stream>>>(kv, wob, w2t);
  out_kernel<<<dim3(N_ / 32, G_, B_), 256, 0, stream>>>(x, pos, wqb, bq, freqs, w2t,
                                                        bo, y);
}
